// Model_21440476742114
// MI455X (gfx1250) — matrix operand order repaired and checked by reading, not compiled, not run on hardware
//
#include <hip/hip_runtime.h>
#include <hip/hip_bf16.h>
#include <math.h>

// ---------------------------------------------------------------------------
// Model sizes (fixed by the reference)
// ---------------------------------------------------------------------------
#define VOCAB 32000
#define EDIM  512
#define HDIM  512
#define LSEQ  64
#define BATCH 32
#define H2    1024
#define SOS   127

typedef __bf16 v16bf __attribute__((ext_vector_type(16)));
typedef __bf16 v8bf  __attribute__((ext_vector_type(8)));
typedef float  v8f   __attribute__((ext_vector_type(8)));

// ---------------------------------------------------------------------------
// bf16 WMMA GEMM, 32x32 output block per wave (2 M-tiles x 2 N-tiles):
//   C[32,N] = A[32,K] * W[N,K]^T + bias
// A row-major [32][K] bf16, W row-major [N][K] bf16 (natural weight layout).
// M fixed at 32; N multiple of 32; K multiple of 32.
// 4 independent accumulators -> back-to-back WMMA issue; each A/B fragment
// reused twice, so W (the streamed matrix) is read exactly once.
// ---------------------------------------------------------------------------
__global__ void gemm_wmma_bf16(const __bf16* __restrict__ A,
                               const __bf16* __restrict__ W,
                               const float*  __restrict__ bias,
                               float* __restrict__ Cf,
                               __bf16* __restrict__ Cb,
                               int N, int K, int ldc, int relu) {
    const int wave = (blockIdx.x * blockDim.x + threadIdx.x) >> 5;
    const int lane = threadIdx.x & 31;
    const int n0 = wave * 32;
    if (n0 >= N) return;   // whole-wave uniform exit (EXEC all-1 for WMMA)

    // A-fragment addressing (16x32 bf16 tile): lane l -> row m=l&15,
    // K half {(l>>4)*16..+15} contiguous
    const int a_kg = (lane >> 4) * 16;
    const __bf16* __restrict__ arow0 = A + (size_t)(lane & 15) * K;          // M-tile 0
    const __bf16* __restrict__ arow1 = A + (size_t)(16 + (lane & 15)) * K;   // M-tile 1
    // B-fragment addressing (32x16 bf16 tile): lane l -> col n=l&15,
    // K half {(l>>4)*16..+15} contiguous
    const int b_ks = (lane >> 4) * 16;
    const __bf16* __restrict__ brow0 = W + (size_t)(n0 + (lane & 15)) * K;       // N-tile 0
    const __bf16* __restrict__ brow1 = W + (size_t)(n0 + 16 + (lane & 15)) * K;  // N-tile 1

    v8f acc00 = {}, acc01 = {}, acc10 = {}, acc11 = {};
#pragma unroll 2
    for (int k0 = 0; k0 < K; k0 += 32) {
        v8bf a0lo = *(const v8bf*)(arow0 + k0 + a_kg);
        v8bf a0hi = *(const v8bf*)(arow0 + k0 + a_kg + 8);
        v8bf a1lo = *(const v8bf*)(arow1 + k0 + a_kg);
        v8bf a1hi = *(const v8bf*)(arow1 + k0 + a_kg + 8);
        v16bf a0 = __builtin_shufflevector(a0lo, a0hi,
                     0,1,2,3,4,5,6,7,8,9,10,11,12,13,14,15);
        v16bf a1 = __builtin_shufflevector(a1lo, a1hi,
                     0,1,2,3,4,5,6,7,8,9,10,11,12,13,14,15);
        v16bf b0 = *(const v16bf*)(brow0 + k0 + b_ks);
        v16bf b1 = *(const v16bf*)(brow1 + k0 + b_ks);
        acc00 = __builtin_amdgcn_wmma_f32_16x16x32_bf16(false, a0, false, b0, (short)0, acc00, false, false);
        acc01 = __builtin_amdgcn_wmma_f32_16x16x32_bf16(false, a0, false, b1, (short)0, acc01, false, false);
        acc10 = __builtin_amdgcn_wmma_f32_16x16x32_bf16(false, a1, false, b0, (short)0, acc10, false, false);
        acc11 = __builtin_amdgcn_wmma_f32_16x16x32_bf16(false, a1, false, b1, (short)0, acc11, false, false);
    }

    // C layout per tile: lane l, VGPR r -> (m = r + 8*(l>>4), n = l&15)
    const int nA = n0 + (lane & 15);
    const int nB = n0 + 16 + (lane & 15);
    const float bvA = bias ? bias[nA] : 0.0f;
    const float bvB = bias ? bias[nB] : 0.0f;
    const int mbase = (lane >> 4) << 3;

#pragma unroll
    for (int r = 0; r < 8; ++r) {
        const int m0 = mbase + r;        // M-tile 0 row
        const int m1 = 16 + mbase + r;   // M-tile 1 row
        float v00 = acc00[r] + bvA;
        float v01 = acc01[r] + bvB;
        float v10 = acc10[r] + bvA;
        float v11 = acc11[r] + bvB;
        if (relu) {
            v00 = fmaxf(v00, 0.0f); v01 = fmaxf(v01, 0.0f);
            v10 = fmaxf(v10, 0.0f); v11 = fmaxf(v11, 0.0f);
        }
        const size_t o00 = (size_t)m0 * ldc + nA;
        const size_t o01 = (size_t)m0 * ldc + nB;
        const size_t o10 = (size_t)m1 * ldc + nA;
        const size_t o11 = (size_t)m1 * ldc + nB;
        if (Cf) { Cf[o00] = v00; Cf[o01] = v01; Cf[o10] = v10; Cf[o11] = v11; }
        if (Cb) {
            Cb[o00] = (__bf16)v00; Cb[o01] = (__bf16)v01;
            Cb[o10] = (__bf16)v10; Cb[o11] = (__bf16)v11;
        }
    }
}

// ---------------------------------------------------------------------------
// Weight prep
// ---------------------------------------------------------------------------
__global__ void k_convert_bf16(const float* __restrict__ in, __bf16* __restrict__ out,
                               long long n) {
    for (long long i = (long long)blockIdx.x * blockDim.x + threadIdx.x; i < n;
         i += (long long)gridDim.x * blockDim.x)
        out[i] = (__bf16)in[i];
}

// out[n][0:K1] = Wa[n][:], out[n][K1:K1+K2] = Wb[n][:]   (fused [Wih|Whh], bf16)
__global__ void k_fuse2_bf16(const float* __restrict__ Wa, const float* __restrict__ Wb,
                             __bf16* __restrict__ out, int N, int K1, int K2) {
    const int K = K1 + K2;
    const long long total = (long long)N * K;
    for (long long i = (long long)blockIdx.x * blockDim.x + threadIdx.x; i < total;
         i += (long long)gridDim.x * blockDim.x) {
        const int n = (int)(i / K), k = (int)(i % K);
        const float v = (k < K1) ? Wa[(size_t)n * K1 + k]
                                 : Wb[(size_t)n * K2 + (k - K1)];
        out[i] = (__bf16)v;
    }
}

// ---------------------------------------------------------------------------
// State init
// ---------------------------------------------------------------------------
__global__ void k_init(float* cf, float* cb, float* h, float* c,
                       __bf16* xf, __bf16* xb, int* tok) {
    const int i = blockIdx.x * blockDim.x + threadIdx.x;   // 0 .. 32*1024-1
    if (i < BATCH * HDIM) { cf[i] = 0.f; cb[i] = 0.f; }
    if (i < BATCH * H2)   { h[i] = 0.f;  c[i] = 0.f; }
    if (i < BATCH * HDIM) {
        const int b = i >> 9, j = i & 511;
        xf[b * 1024 + 512 + j] = (__bf16)0.f;
        xb[b * 1024 + 512 + j] = (__bf16)0.f;
    }
    if (i < BATCH) tok[i] = SOS;
}

// ---------------------------------------------------------------------------
// Encoder per-step kernels
// ---------------------------------------------------------------------------
__global__ void k_prep_enc(const int* __restrict__ x, const float* __restrict__ enc_embed,
                           __bf16* __restrict__ xf, __bf16* __restrict__ xb, int t) {
    const int i = blockIdx.x * blockDim.x + threadIdx.x;   // 32*512
    if (i >= BATCH * EDIM) return;
    const int b = i >> 9, j = i & 511;
    const int row = x[b * LSEQ + t];
    const __bf16 v = (__bf16)enc_embed[(size_t)row * EDIM + j];
    xf[b * 1024 + j] = v;
    xb[b * 1024 + j] = v;
}

__device__ __forceinline__ float sigm(float x) { return 1.0f / (1.0f + expf(-x)); }

__global__ void k_lstm_enc(const float* __restrict__ gf, const float* __restrict__ gb,
                           float* __restrict__ cf, float* __restrict__ cb,
                           __bf16* __restrict__ xf, __bf16* __restrict__ xb,
                           float* __restrict__ enc_outs, int t) {
    const int i = blockIdx.x * blockDim.x + threadIdx.x;   // 2*32*512
    if (i >= 2 * BATCH * HDIM) return;
    const int dir = (i >= BATCH * HDIM);
    const int r = i - dir * BATCH * HDIM;
    const int b = r >> 9, j = r & 511;
    const float* g = dir ? gb : gf;
    float* cs = dir ? cb : cf;
    const float ig = g[b * 2048 + j];
    const float fg = g[b * 2048 + 512 + j];
    const float gg = g[b * 2048 + 1024 + j];
    const float og = g[b * 2048 + 1536 + j];
    const float c2 = sigm(fg) * cs[r] + sigm(ig) * tanhf(gg);
    const float hv = sigm(og) * tanhf(c2);
    cs[r] = c2;
    (dir ? xb : xf)[b * 1024 + 512 + j] = (__bf16)hv;
    enc_outs[(size_t)b * LSEQ * 1024 + (size_t)t * 1024 + dir * 512 + j] = hv;
}

// ---------------------------------------------------------------------------
// Decoder per-step kernels
// ---------------------------------------------------------------------------
// xa = [e | h] (attn input), xc[0:H2] = e (comb input), xd[H2:2H2] = h (lstm input)
__global__ void k_prep_dec(const int* __restrict__ tok, const float* __restrict__ dec_embed,
                           const float* __restrict__ h,
                           __bf16* __restrict__ xa, __bf16* __restrict__ xc,
                           __bf16* __restrict__ xd) {
    const int i = blockIdx.x * blockDim.x + threadIdx.x;   // 32*1024
    if (i >= BATCH * H2) return;
    const int b = i >> 10, j = i & 1023;
    const __bf16 eb = (__bf16)dec_embed[(size_t)tok[b] * H2 + j];
    xa[b * 2048 + j] = eb;
    xc[b * 2048 + j] = eb;
    const __bf16 hb = (__bf16)h[i];
    xa[b * 2048 + 1024 + j] = hb;
    xd[b * 2048 + 1024 + j] = hb;
}

// softmax over L=64 scores, then ctx[b,:] = sum_l aw[l] * enc_outs[b,l,:] -> bf16 into xc hi half
__global__ void k_softmax_ctx(const float* __restrict__ scores,
                              const float* __restrict__ enc_outs,
                              __bf16* __restrict__ xc) {
    const int b = blockIdx.x;
    const int tid = threadIdx.x;
    __shared__ float aw[LSEQ];
    __shared__ float raw[LSEQ];
    if (tid < LSEQ) raw[tid] = scores[b * LSEQ + tid];
    __syncthreads();
    if (tid == 0) {
        float mx = raw[0];
        for (int l = 1; l < LSEQ; ++l) mx = fmaxf(mx, raw[l]);
        float s = 0.f;
        for (int l = 0; l < LSEQ; ++l) { float e = expf(raw[l] - mx); aw[l] = e; s += e; }
        const float inv = 1.0f / s;
        for (int l = 0; l < LSEQ; ++l) aw[l] *= inv;
    }
    __syncthreads();
    const float* eo = enc_outs + (size_t)b * LSEQ * 1024;
    for (int d = tid; d < 1024; d += blockDim.x) {
        float acc = 0.f;
#pragma unroll 8
        for (int l = 0; l < LSEQ; ++l) acc += aw[l] * eo[l * 1024 + d];
        xc[b * 2048 + 1024 + d] = (__bf16)acc;
    }
}

__global__ void k_lstm_dec(const float* __restrict__ gates, float* __restrict__ h,
                           float* __restrict__ c, __bf16* __restrict__ h_bf) {
    const int i = blockIdx.x * blockDim.x + threadIdx.x;   // 32*1024
    if (i >= BATCH * H2) return;
    const int b = i >> 10, j = i & 1023;
    const float ig = gates[b * 4096 + j];
    const float fg = gates[b * 4096 + 1024 + j];
    const float gg = gates[b * 4096 + 2048 + j];
    const float og = gates[b * 4096 + 3072 + j];
    const float c2 = sigm(fg) * c[i] + sigm(ig) * tanhf(gg);
    const float hv = sigm(og) * tanhf(c2);
    c[i] = c2;
    h[i] = hv;
    h_bf[i] = (__bf16)hv;
}

// log_softmax over V=32000 + greedy argmax (first-max tie-break), writes d_out[b][t][:]
__global__ void k_logsoftmax_argmax(const float* __restrict__ logits,
                                    float* __restrict__ out, int* __restrict__ tok, int t) {
    const int b = blockIdx.x;
    const int tid = threadIdx.x;                 // 256 threads
    const float* row = logits + (size_t)b * VOCAB;
    __shared__ float smax[256];
    __shared__ int   sidx[256];
    __shared__ float ssum[256];

    float mx = -INFINITY; int mi = 0x7fffffff;
    for (int v = tid; v < VOCAB; v += 256) {
        const float x = row[v];
        if (x > mx) { mx = x; mi = v; }
    }
    smax[tid] = mx; sidx[tid] = mi;
    __syncthreads();
    for (int s = 128; s > 0; s >>= 1) {
        if (tid < s) {
            if (smax[tid + s] > smax[tid] ||
                (smax[tid + s] == smax[tid] && sidx[tid + s] < sidx[tid])) {
                smax[tid] = smax[tid + s]; sidx[tid] = sidx[tid + s];
            }
        }
        __syncthreads();
    }
    const float gmax = smax[0];

    float ss = 0.f;
    for (int v = tid; v < VOCAB; v += 256) ss += expf(row[v] - gmax);
    ssum[tid] = ss;
    __syncthreads();
    for (int s = 128; s > 0; s >>= 1) {
        if (tid < s) ssum[tid] += ssum[tid + s];
        __syncthreads();
    }
    const float lse = gmax + logf(ssum[0]);

    float* orow = out + ((size_t)b * LSEQ + (size_t)t) * VOCAB;
    for (int v = tid; v < VOCAB; v += 256) orow[v] = row[v] - lse;
    if (tid == 0) tok[b] = sidx[0];
}

// ---------------------------------------------------------------------------
// Host orchestration
// ---------------------------------------------------------------------------
extern "C" void kernel_launch(void* const* d_in, const int* in_sizes, int n_in,
                              void* d_out, int out_size, void* d_ws, size_t ws_size,
                              hipStream_t stream) {
    (void)in_sizes; (void)n_in; (void)out_size; (void)ws_size;

    const int*   x         = (const int*)  d_in[0];
    const float* enc_embed = (const float*)d_in[1];
    const float* ef_Wih    = (const float*)d_in[2];
    const float* ef_Whh    = (const float*)d_in[3];
    const float* ef_b      = (const float*)d_in[4];
    const float* eb_Wih    = (const float*)d_in[5];
    const float* eb_Whh    = (const float*)d_in[6];
    const float* eb_b      = (const float*)d_in[7];
    const float* dec_embed = (const float*)d_in[8];
    const float* attn_W    = (const float*)d_in[9];
    const float* attn_b    = (const float*)d_in[10];
    const float* comb_W    = (const float*)d_in[11];
    const float* comb_b    = (const float*)d_in[12];
    const float* d_Wih     = (const float*)d_in[13];
    const float* d_Whh     = (const float*)d_in[14];
    const float* d_b       = (const float*)d_in[15];
    const float* out_W     = (const float*)d_in[16];
    const float* out_b     = (const float*)d_in[17];
    float* out = (float*)d_out;

    // -------- workspace carve-up (256B aligned slabs) --------
    char* ws = (char*)d_ws;
    size_t off = 0;
    auto alloc = [&](size_t bytes) -> void* {
        void* p = ws + off;
        off = (off + bytes + 255) & ~(size_t)255;
        return p;
    };
    __bf16* outW_bf  = (__bf16*)alloc((size_t)VOCAB * H2 * 2);     // 65.5 MB (L2-resident)
    __bf16* decW_bf  = (__bf16*)alloc((size_t)4 * H2 * 2 * H2 * 2);// 16 MB  [d_Wih|d_Whh]
    __bf16* encWf_bf = (__bf16*)alloc((size_t)2048 * 1024 * 2);    // [ef_Wih|ef_Whh]
    __bf16* encWb_bf = (__bf16*)alloc((size_t)2048 * 1024 * 2);
    __bf16* combW_bf = (__bf16*)alloc((size_t)H2 * 2048 * 2);
    __bf16* attnW_bf = (__bf16*)alloc((size_t)LSEQ * 2048 * 2);
    float*  enc_outs = (float*) alloc((size_t)BATCH * LSEQ * 1024 * 4);   // 8 MB
    float*  gates_f  = (float*) alloc((size_t)BATCH * 2048 * 4);
    float*  gates_b  = (float*) alloc((size_t)BATCH * 2048 * 4);
    float*  gates_d  = (float*) alloc((size_t)BATCH * 4096 * 4);
    float*  logits   = (float*) alloc((size_t)BATCH * VOCAB * 4);         // 4 MB
    __bf16* xf       = (__bf16*)alloc((size_t)BATCH * 1024 * 2);
    __bf16* xb       = (__bf16*)alloc((size_t)BATCH * 1024 * 2);
    __bf16* xa       = (__bf16*)alloc((size_t)BATCH * 2048 * 2);
    __bf16* xc       = (__bf16*)alloc((size_t)BATCH * 2048 * 2);
    __bf16* xd       = (__bf16*)alloc((size_t)BATCH * 2048 * 2);
    __bf16* h_bf     = (__bf16*)alloc((size_t)BATCH * H2 * 2);
    float*  cf       = (float*) alloc((size_t)BATCH * HDIM * 4);
    float*  cb       = (float*) alloc((size_t)BATCH * HDIM * 4);
    float*  hd       = (float*) alloc((size_t)BATCH * H2 * 4);
    float*  cd       = (float*) alloc((size_t)BATCH * H2 * 4);
    float*  scores   = (float*) alloc((size_t)BATCH * LSEQ * 4);
    int*    tok      = (int*)   alloc((size_t)BATCH * 4);

    // M is always 32; one wave per 32x32 output block
    auto gemm = [&](const __bf16* A, const __bf16* W, const float* bias,
                    float* Cf, __bf16* Cb, int N, int K, int ldc, int relu) {
        const int waves = N / 32;
        const int threads = 128;                      // 4 waves / block
        const int blocks = (waves + 3) / 4;
        gemm_wmma_bf16<<<blocks, threads, 0, stream>>>(A, W, bias, Cf, Cb, N, K, ldc, relu);
    };

    // -------- one-time weight prep (inside graph; deterministic) --------
    k_convert_bf16<<<4096, 256, 0, stream>>>(out_W, outW_bf, (long long)VOCAB * H2);
    k_convert_bf16<<<1024, 256, 0, stream>>>(comb_W, combW_bf, (long long)H2 * 2048);
    k_convert_bf16<<<256,  256, 0, stream>>>(attn_W, attnW_bf, (long long)LSEQ * 2048);
    k_fuse2_bf16<<<2048, 256, 0, stream>>>(ef_Wih, ef_Whh, encWf_bf, 2048, 512, 512);
    k_fuse2_bf16<<<2048, 256, 0, stream>>>(eb_Wih, eb_Whh, encWb_bf, 2048, 512, 512);
    k_fuse2_bf16<<<4096, 256, 0, stream>>>(d_Wih, d_Whh, decW_bf, 4096, 1024, 1024);

    k_init<<<128, 256, 0, stream>>>(cf, cb, hd, cd, xf, xb, tok);

    // -------- encoder: 64 timesteps, fused [x|h] @ [Wih|Whh]^T per direction --------
    for (int t = 0; t < LSEQ; ++t) {
        k_prep_enc<<<64, 256, 0, stream>>>(x, enc_embed, xf, xb, t);
        gemm(xf, encWf_bf, ef_b, gates_f, nullptr, 2048, 1024, 2048, 0);
        gemm(xb, encWb_bf, eb_b, gates_b, nullptr, 2048, 1024, 2048, 0);
        k_lstm_enc<<<128, 256, 0, stream>>>(gates_f, gates_b, cf, cb, xf, xb, enc_outs, t);
    }

    // -------- decoder: 64 timesteps with attention + greedy feedback --------
    for (int t = 0; t < LSEQ; ++t) {
        k_prep_dec<<<128, 256, 0, stream>>>(tok, dec_embed, hd, xa, xc, xd);
        // attention scores: [e|h] (32x2048) @ attn_W^T (2048->64)
        gemm(xa, attnW_bf, attn_b, scores, nullptr, LSEQ, 2048, LSEQ, 0);
        k_softmax_ctx<<<BATCH, 256, 0, stream>>>(scores, enc_outs, xc);
        // comb = relu([e|ctx] @ comb_W^T + b) -> bf16 into xd[:,0:1024]
        gemm(xc, combW_bf, comb_b, nullptr, xd, H2, 2048, 2048, 1);
        // gates = [comb|h] @ [d_Wih|d_Whh]^T + d_b
        gemm(xd, decW_bf, d_b, gates_d, nullptr, 4096, 2048, 4096, 0);
        k_lstm_dec<<<128, 256, 0, stream>>>(gates_d, hd, cd, h_bf);
        // output projection: h (32x1024) @ out_W^T (1024->32000)  [dominant GEMM]
        gemm(h_bf, outW_bf, out_b, logits, nullptr, VOCAB, 1024, VOCAB, 0);
        k_logsoftmax_argmax<<<BATCH, 256, 0, stream>>>(logits, out, tok, t);
    }
}
